// MultiHeadAttention_3324304687649
// MI455X (gfx1250) — matrix operand order repaired and checked by reading, not compiled, not run on hardware
//
#include <hip/hip_runtime.h>
#include <hip/hip_bf16.h>

// ---------------------------------------------------------------------------
// CDNA5 (gfx1250, wave32) fused multi-head attention.
// All GEMMs via V_WMMA_F32_16X16X32_F16; K/V tiles staged to LDS with
// double-buffered GLOBAL_LOAD_ASYNC_TO_LDS_B128 (ASYNCcnt); GEMM loops are
// software-pipelined; mask pre-packed to bits with wave32 __ballot.
// ---------------------------------------------------------------------------

typedef __attribute__((ext_vector_type(16))) _Float16 v16h;
typedef __attribute__((ext_vector_type(8)))  _Float16 h8;
typedef __attribute__((ext_vector_type(4)))  _Float16 h4;
typedef __attribute__((ext_vector_type(8)))  float    v8f;
typedef __attribute__((ext_vector_type(4)))  float    f4;
typedef __attribute__((ext_vector_type(4)))  int      i4;

#define DEV __device__ __forceinline__

static constexpr int   D_MODEL = 1024;
static constexpr int   S_LEN   = 2048;
static constexpr int   NHEAD   = 16;
static constexpr int   DEPTH   = 64;
static constexpr float NORM    = 0.125f;    // 1/sqrt(64)
static constexpr int   ROWS    = 2 * S_LEN; // B*S = 4096 flat rows

#if __has_builtin(__builtin_amdgcn_global_load_async_to_lds_b128)
#define HAVE_ASYNC_LDS 1
typedef __attribute__((address_space(1))) i4 gi4;   // global int4
typedef __attribute__((address_space(3))) i4 li4;   // LDS int4
#else
#define HAVE_ASYNC_LDS 0
#endif

DEV void wait_async_all() {
#if HAVE_ASYNC_LDS
#if __has_builtin(__builtin_amdgcn_s_wait_asynccnt)
  __builtin_amdgcn_s_wait_asynccnt(0);
#else
  asm volatile("s_wait_asynccnt 0" ::: "memory");
#endif
#endif
}

DEV v8f wmma16(v16h a, v16h b, v8f c) {
  // (neg_a, A, neg_b, B, c_mod, C, reuse_a, reuse_b)
  return __builtin_amdgcn_wmma_f32_16x16x32_f16(false, a, false, b,
                                                (short)0, c, false, false);
}

// A fragment (16x32 f16, M x K): lane L: m = L&15; halves i<8 -> K=k0+8*hi+i,
// halves i>=8 -> K = k0+16+8*hi+(i-8).  Two contiguous 16B chunks.
DEV v16h load_a_f16(const _Float16* rowptr, int k0, int hi) {
  h8 x = *(const h8*)(rowptr + k0 + 8 * hi);
  h8 y = *(const h8*)(rowptr + k0 + 8 * hi + 16);
  v16h a;
#pragma unroll
  for (int i = 0; i < 8; ++i) { a[i] = x[i]; a[i + 8] = y[i]; }
  return a;
}

// B fragment (32x16 f16, K x N): lane L: n = L&15; halves i<8 -> K=k0+8*hi+i,
// halves i>=8 -> K = k0+16+8*hi+(i-8).  Two contiguous 16B chunks.
DEV v16h load_b_f16(const _Float16* rowptr, int k0, int hi) {
  const h8* p = (const h8*)(rowptr + k0 + 8 * hi);
  h8 x = p[0], y = p[2];
  v16h b;
#pragma unroll
  for (int i = 0; i < 8; ++i) { b[i] = x[i]; b[i + 8] = y[i]; }
  return b;
}

DEV float grp_max16(float v) {
#pragma unroll
  for (int m = 1; m < 16; m <<= 1) v = fmaxf(v, __shfl_xor(v, m, 32));
  return v;
}
DEV float grp_sum16(float v) {
#pragma unroll
  for (int m = 1; m < 16; m <<= 1) v += __shfl_xor(v, m, 32);
  return v;
}

// ---------------------------------------------------------------------------
// Kernel 0a: one-pass fp32 -> fp16 conversion (vector of 4 per thread).
// ---------------------------------------------------------------------------
__global__ void __launch_bounds__(256)
cvt_kernel(const float* __restrict__ src, _Float16* __restrict__ dst, int n4) {
  const int i = blockIdx.x * blockDim.x + threadIdx.x;
  if (i < n4) {
    f4 x = ((const f4*)src)[i];
    h4 y;
#pragma unroll
    for (int j = 0; j < 4; ++j) y[j] = (_Float16)x[j];
    ((h4*)dst)[i] = y;
  }
}

// ---------------------------------------------------------------------------
// Kernel 0b: pack int32 mask to 1 bit/element with wave32 ballot.
// bit i of mbits[w] == (mask[w*32+i] != 0)
// ---------------------------------------------------------------------------
__global__ void __launch_bounds__(256)
pack_mask_kernel(const int* __restrict__ mask, unsigned* __restrict__ mbits) {
  const int t = blockIdx.x * blockDim.x + threadIdx.x;
  const unsigned long long bal = __ballot(mask[t] != 0);
  if ((threadIdx.x & 31) == 0) mbits[t >> 5] = (unsigned)bal;
}

// ---------------------------------------------------------------------------
// Kernel 1: projection  Y = X @ W^T + bias, fp16 in, fp16 out.
// mode 0: Y*NORM -> qh [b,h,s,d] ; mode 1: Y -> kh [b,h,s,d]
// mode 2: Y -> vt [b,h,d,s] (transposed for the P*V stage)
// One wave = 16x64 output tile; software-pipelined b128 loads + WMMA.
// ---------------------------------------------------------------------------
__global__ void __launch_bounds__(256)
proj_kernel(const _Float16* __restrict__ X, const _Float16* __restrict__ W,
            const float* __restrict__ bias, _Float16* __restrict__ out,
            int mode) {
  const int wave = blockIdx.x * (blockDim.x >> 5) + (threadIdx.x >> 5);
  const int lane = threadIdx.x & 31;
  const int hi   = lane >> 4;
  const int ln   = lane & 15;
  const int rt = wave >> 4;        // 256 row tiles of 16
  const int cg = wave & 15;        // 16 col groups of 64
  const int row0 = rt * 16;

  v8f acc[4];
#pragma unroll
  for (int j = 0; j < 4; ++j) {
    const float bv = bias[cg * 64 + j * 16 + ln];
#pragma unroll
    for (int r = 0; r < 8; ++r) acc[j][r] = bv;
  }

  const _Float16* xrow = X + (size_t)(row0 + ln) * D_MODEL;
  const _Float16* wrow0 = W + (size_t)(cg * 64 + ln) * D_MODEL;

  // Software pipeline: fragments for step k+1 load during step k's WMMAs.
  v16h a = load_a_f16(xrow, 0, hi);
  v16h b0 = load_b_f16(wrow0 + (size_t)(0 * 16) * D_MODEL, 0, hi);
  v16h b1 = load_b_f16(wrow0 + (size_t)(1 * 16) * D_MODEL, 0, hi);
  v16h b2 = load_b_f16(wrow0 + (size_t)(2 * 16) * D_MODEL, 0, hi);
  v16h b3 = load_b_f16(wrow0 + (size_t)(3 * 16) * D_MODEL, 0, hi);
  for (int kk = 0; kk < D_MODEL - 32; kk += 32) {
    const int kn = kk + 32;
    v16h an  = load_a_f16(xrow, kn, hi);
    v16h bn0 = load_b_f16(wrow0 + (size_t)(0 * 16) * D_MODEL, kn, hi);
    v16h bn1 = load_b_f16(wrow0 + (size_t)(1 * 16) * D_MODEL, kn, hi);
    v16h bn2 = load_b_f16(wrow0 + (size_t)(2 * 16) * D_MODEL, kn, hi);
    v16h bn3 = load_b_f16(wrow0 + (size_t)(3 * 16) * D_MODEL, kn, hi);
    acc[0] = wmma16(a, b0, acc[0]);
    acc[1] = wmma16(a, b1, acc[1]);
    acc[2] = wmma16(a, b2, acc[2]);
    acc[3] = wmma16(a, b3, acc[3]);
    a = an; b0 = bn0; b1 = bn1; b2 = bn2; b3 = bn3;
  }
  acc[0] = wmma16(a, b0, acc[0]);
  acc[1] = wmma16(a, b1, acc[1]);
  acc[2] = wmma16(a, b2, acc[2]);
  acc[3] = wmma16(a, b3, acc[3]);

#pragma unroll
  for (int j = 0; j < 4; ++j) {
#pragma unroll
    for (int r = 0; r < 8; ++r) {
      const int gr = row0 + r + 8 * hi;      // 0..4095
      const int bb = gr >> 11;
      const int s  = gr & (S_LEN - 1);
      const int o  = cg * 64 + j * 16 + ln;
      const int hh = o >> 6;
      const int d  = o & 63;
      float val = acc[j][r];
      if (mode == 0) val *= NORM;
      size_t idx;
      if (mode == 2)
        idx = (((size_t)(bb * NHEAD + hh) * DEPTH + d) * S_LEN + s);
      else
        idx = (((size_t)(bb * NHEAD + hh) * S_LEN + s) * DEPTH + d);
      out[idx] = (_Float16)val;
    }
  }
}

// ---------------------------------------------------------------------------
// Kernel 2: streaming attention. One block = 8 waves covering 8 query tiles
// of the SAME (b,h). Per 32-key iteration the block stages shared K (32x64)
// and V^T (64x32) tiles into double-buffered LDS with async global->LDS
// copies, overlapping the next fetch with this tile's 8 WMMAs.
// ---------------------------------------------------------------------------
__global__ void __launch_bounds__(256)
attn_kernel(const _Float16* __restrict__ qh, const _Float16* __restrict__ kh,
            const _Float16* __restrict__ vt,
            const unsigned* __restrict__ mbits, _Float16* __restrict__ ctxh) {
  __shared__ __align__(16) _Float16 ktile[2][32 * 64];  // 2 x 4 KB
  __shared__ __align__(16) _Float16 vtile[2][64 * 32];  // 2 x 4 KB
  __shared__ __align__(16) _Float16 pt[8][16 * 32];     // 8 x 1 KB

  const int tid   = threadIdx.x;
  const int lwave = tid >> 5;
  const int lane  = tid & 31;
  const int hi    = lane >> 4;
  const int ln    = lane & 15;
  const int bh = blockIdx.x >> 4;               // 0..31  (b*16+h)
  const int qt = (blockIdx.x & 15) * 8 + lwave; // 128 query tiles per head
  const int b  = bh >> 4;
  const int h  = bh & 15;

  const _Float16* kbase = kh + (size_t)bh * S_LEN * DEPTH;
  const _Float16* vbase = vt + (size_t)bh * DEPTH * S_LEN;
  const unsigned* mb    = mbits + (size_t)b * S_LEN * (S_LEN / 32);
  _Float16* myp = &pt[lwave][0];

  // Stage one 32-key step: K tile row = tid/8, 16B chunk = tid%8;
  // V tile row (d) = tid/4, 16B chunk = tid%4.  2 async b128 per thread.
  auto stage = [&](int kb, int buf) {
    const _Float16* gk = kbase + (size_t)(kb + (tid >> 3)) * DEPTH + (tid & 7) * 8;
    _Float16*       lk = &ktile[buf][(tid >> 3) * DEPTH + (tid & 7) * 8];
    const _Float16* gv = vbase + (size_t)(tid >> 2) * S_LEN + kb + (tid & 3) * 8;
    _Float16*       lv = &vtile[buf][(tid >> 2) * 32 + (tid & 3) * 8];
#if HAVE_ASYNC_LDS
    __builtin_amdgcn_global_load_async_to_lds_b128((gi4*)gk, (li4*)lk, 0, 0);
    __builtin_amdgcn_global_load_async_to_lds_b128((gi4*)gv, (li4*)lv, 0, 0);
#else
    *(h8*)lk = *(const h8*)gk;
    *(h8*)lv = *(const h8*)gv;
#endif
  };

  // Q fragments for this 16-row tile (K = depth = 64 -> two 16x32 frags)
  const _Float16* qrow = qh + ((size_t)bh * S_LEN + qt * 16 + ln) * DEPTH;
  const v16h aq0 = load_a_f16(qrow, 0, hi);
  const v16h aq1 = load_a_f16(qrow, 32, hi);

  float mrun[8], lrun[8];
  v8f ctx[4];
#pragma unroll
  for (int r = 0; r < 8; ++r) { mrun[r] = -3.0e38f; lrun[r] = 0.0f; }
#pragma unroll
  for (int j = 0; j < 4; ++j)
#pragma unroll
    for (int r = 0; r < 8; ++r) ctx[j][r] = 0.0f;

  // Prime the pipeline.
  stage(0, 0);
  wait_async_all();
  __syncthreads();

  int cur = 0;
  for (int kb = 0; kb < S_LEN; kb += 32) {
    if (kb + 32 < S_LEN) stage(kb + 32, cur ^ 1);  // overlap with compute

    const _Float16* kc = &ktile[cur][0];
    const _Float16* vc = &vtile[cur][0];

    // ---- scores: S = (q*NORM) @ k^T, two 16x16 tiles, 4 WMMAs ----
    v16h bk00 = load_b_f16(kc + (size_t)ln * DEPTH, 0, hi);
    v16h bk10 = load_b_f16(kc + (size_t)ln * DEPTH, 32, hi);
    v16h bk01 = load_b_f16(kc + (size_t)(16 + ln) * DEPTH, 0, hi);
    v16h bk11 = load_b_f16(kc + (size_t)(16 + ln) * DEPTH, 32, hi);
    v8f zc = {};
    v8f s0 = wmma16(aq1, bk10, wmma16(aq0, bk00, zc));
    v8f s1 = wmma16(aq1, bk11, wmma16(aq0, bk01, zc));

    // ---- masking via packed bits (uniform dword per 16-lane group) ----
    // faithful to reference: scores + where(mask==0, -1e9, scores)
#pragma unroll
    for (int r = 0; r < 8; ++r) {
      const int qr = qt * 16 + r + 8 * hi;
      const unsigned w = mb[(size_t)qr * (S_LEN / 32) + (kb >> 5)];
      const float v0 = s0[r];
      const float v1 = s1[r];
      s0[r] = ((w >> ln) & 1u)        ? (v0 + v0) : (v0 - 1.0e9f);
      s1[r] = ((w >> (16 + ln)) & 1u) ? (v1 + v1) : (v1 - 1.0e9f);
    }

    // ---- online softmax update (row = fixed r within 16-lane group) ----
#pragma unroll
    for (int r = 0; r < 8; ++r) {
      float tmax = grp_max16(fmaxf(s0[r], s1[r]));
      float mnew = fmaxf(mrun[r], tmax);
      float scale = __expf(mrun[r] - mnew);
      float p0 = __expf(s0[r] - mnew);
      float p1 = __expf(s1[r] - mnew);
      s0[r] = p0;
      s1[r] = p1;
      lrun[r] = lrun[r] * scale + grp_sum16(p0 + p1);
      mrun[r] = mnew;
#pragma unroll
      for (int j = 0; j < 4; ++j) ctx[j][r] *= scale;
    }

    // ---- transpose P (C-frag f32 -> A-frag f16) via per-wave LDS tile ----
#pragma unroll
    for (int r = 0; r < 8; ++r) {
      const int row = r + 8 * hi;
      myp[row * 32 + ln]      = (_Float16)s0[r];
      myp[row * 32 + 16 + ln] = (_Float16)s1[r];
    }
    asm volatile("s_wait_dscnt 0" ::: "memory");
    h8 px = *(const h8*)(myp + ln * 32 + 8 * hi);
    h8 py = *(const h8*)(myp + ln * 32 + 8 * hi + 16);
    v16h ap;
#pragma unroll
    for (int i = 0; i < 8; ++i) { ap[i] = px[i]; ap[i + 8] = py[i]; }

    // ---- ctx(16x64) += P(16x32) @ V(32x64), V tile pre-transposed [d][s] ----
#pragma unroll
    for (int j = 0; j < 4; ++j) {
      v16h bv = load_b_f16(vc + (size_t)(j * 16 + ln) * 32, 0, hi);
      ctx[j] = wmma16(ap, bv, ctx[j]);
    }

    wait_async_all();   // next-tile async copies have landed
    __syncthreads();    // every wave done reading cur, data for cur^1 visible
    cur ^= 1;
  }

  // ---- normalize and store fp16 ctx in head-concat layout [b,s,1024] ----
#pragma unroll
  for (int r = 0; r < 8; ++r) {
    const float inv = 1.0f / lrun[r];
    const int row = qt * 16 + r + 8 * hi;
#pragma unroll
    for (int j = 0; j < 4; ++j) {
      const float val = ctx[j][r] * inv;
      ctxh[((size_t)b * S_LEN + row) * D_MODEL + h * DEPTH + j * 16 + ln] =
          (_Float16)val;
    }
  }
}

// ---------------------------------------------------------------------------
// Kernel 3: output projection  out = ctx @ Wo^T + bo  (fp16 in, fp32 out)
// ---------------------------------------------------------------------------
__global__ void __launch_bounds__(256)
outproj_kernel(const _Float16* __restrict__ ctxh,
               const _Float16* __restrict__ Wo,
               const float* __restrict__ bo, float* __restrict__ out) {
  const int wave = blockIdx.x * (blockDim.x >> 5) + (threadIdx.x >> 5);
  const int lane = threadIdx.x & 31;
  const int hi   = lane >> 4;
  const int ln   = lane & 15;
  const int rt = wave >> 4;
  const int cg = wave & 15;
  const int row0 = rt * 16;

  v8f acc[4];
#pragma unroll
  for (int j = 0; j < 4; ++j) {
    const float bv = bo[cg * 64 + j * 16 + ln];
#pragma unroll
    for (int r = 0; r < 8; ++r) acc[j][r] = bv;
  }

  const _Float16* arow = ctxh + (size_t)(row0 + ln) * D_MODEL;
  const _Float16* wrow0 = Wo + (size_t)(cg * 64 + ln) * D_MODEL;

  v16h a = load_a_f16(arow, 0, hi);
  v16h b0 = load_b_f16(wrow0 + (size_t)(0 * 16) * D_MODEL, 0, hi);
  v16h b1 = load_b_f16(wrow0 + (size_t)(1 * 16) * D_MODEL, 0, hi);
  v16h b2 = load_b_f16(wrow0 + (size_t)(2 * 16) * D_MODEL, 0, hi);
  v16h b3 = load_b_f16(wrow0 + (size_t)(3 * 16) * D_MODEL, 0, hi);
  for (int kk = 0; kk < D_MODEL - 32; kk += 32) {
    const int kn = kk + 32;
    v16h an  = load_a_f16(arow, kn, hi);
    v16h bn0 = load_b_f16(wrow0 + (size_t)(0 * 16) * D_MODEL, kn, hi);
    v16h bn1 = load_b_f16(wrow0 + (size_t)(1 * 16) * D_MODEL, kn, hi);
    v16h bn2 = load_b_f16(wrow0 + (size_t)(2 * 16) * D_MODEL, kn, hi);
    v16h bn3 = load_b_f16(wrow0 + (size_t)(3 * 16) * D_MODEL, kn, hi);
    acc[0] = wmma16(a, b0, acc[0]);
    acc[1] = wmma16(a, b1, acc[1]);
    acc[2] = wmma16(a, b2, acc[2]);
    acc[3] = wmma16(a, b3, acc[3]);
    a = an; b0 = bn0; b1 = bn1; b2 = bn2; b3 = bn3;
  }
  acc[0] = wmma16(a, b0, acc[0]);
  acc[1] = wmma16(a, b1, acc[1]);
  acc[2] = wmma16(a, b2, acc[2]);
  acc[3] = wmma16(a, b3, acc[3]);

#pragma unroll
  for (int j = 0; j < 4; ++j) {
#pragma unroll
    for (int r = 0; r < 8; ++r) {
      const int gr = row0 + r + 8 * hi;
      const int o  = cg * 64 + j * 16 + ln;
      out[(size_t)gr * D_MODEL + o] = acc[j][r];
    }
  }
}

// ---------------------------------------------------------------------------
extern "C" void kernel_launch(void* const* d_in, const int* in_sizes, int n_in,
                              void* d_out, int out_size, void* d_ws,
                              size_t ws_size, hipStream_t stream) {
  const float* Q    = (const float*)d_in[0];
  const float* K    = (const float*)d_in[1];
  const float* V    = (const float*)d_in[2];
  const int*   mask = (const int*)d_in[3];
  const float* Wq   = (const float*)d_in[4];
  const float* bq   = (const float*)d_in[5];
  const float* Wk   = (const float*)d_in[6];
  const float* bk   = (const float*)d_in[7];
  const float* Wv   = (const float*)d_in[8];
  const float* bv   = (const float*)d_in[9];
  const float* Wo   = (const float*)d_in[10];
  const float* bo   = (const float*)d_in[11];
  float* out = (float*)d_out;

  // Workspace layout, ~65 MB total:
  //   Qh,Kh,Vh        : converted inputs   (3 x 4,194,304 f16)
  //   Wqh,Wkh,Wvh,Woh : converted weights  (4 x 1,048,576 f16)
  //   qh,kh,vt,ctxh   : intermediates      (4 x 4,194,304 f16)
  //   mbits           : packed mask        (262,144 u32 = 1 MB)
  const size_t NIN = (size_t)ROWS * D_MODEL;
  const size_t NW  = (size_t)D_MODEL * D_MODEL;
  _Float16* Qh   = (_Float16*)d_ws;
  _Float16* Kh   = Qh + NIN;
  _Float16* Vh   = Kh + NIN;
  _Float16* Wqh  = Vh + NIN;
  _Float16* Wkh  = Wqh + NW;
  _Float16* Wvh  = Wkh + NW;
  _Float16* Woh  = Wvh + NW;
  _Float16* qh   = Woh + NW;
  _Float16* kh   = qh + NIN;
  _Float16* vtb  = kh + NIN;
  _Float16* ctxh = vtb + NIN;
  unsigned* mbits = (unsigned*)(ctxh + NIN);

  const dim3 blk(256);
  const int  gcIn = (int)(NIN / 4 / 256);  // 4096 blocks
  const int  gcW  = (int)(NW / 4 / 256);   // 1024 blocks
  const int  gcM  = 2 * S_LEN * S_LEN / 256;  // 32768 blocks (1 int/thread)

  cvt_kernel<<<gcIn, blk, 0, stream>>>(Q, Qh, (int)(NIN / 4));
  cvt_kernel<<<gcIn, blk, 0, stream>>>(K, Kh, (int)(NIN / 4));
  cvt_kernel<<<gcIn, blk, 0, stream>>>(V, Vh, (int)(NIN / 4));
  cvt_kernel<<<gcW,  blk, 0, stream>>>(Wq, Wqh, (int)(NW / 4));
  cvt_kernel<<<gcW,  blk, 0, stream>>>(Wk, Wkh, (int)(NW / 4));
  cvt_kernel<<<gcW,  blk, 0, stream>>>(Wv, Wvh, (int)(NW / 4));
  cvt_kernel<<<gcW,  blk, 0, stream>>>(Wo, Woh, (int)(NW / 4));
  pack_mask_kernel<<<gcM, blk, 0, stream>>>(mask, mbits);

  const dim3 grd(512);  // 4096 waves per GEMM-shaped kernel
  proj_kernel<<<grd, blk, 0, stream>>>(Qh, Wqh, bq, qh, 0);
  proj_kernel<<<grd, blk, 0, stream>>>(Kh, Wkh, bk, kh, 1);
  proj_kernel<<<grd, blk, 0, stream>>>(Vh, Wvh, bv, vtb, 2);
  attn_kernel<<<grd, blk, 0, stream>>>(qh, kh, vtb, mbits, ctxh);
  outproj_kernel<<<grd, blk, 0, stream>>>(ctxh, Woh, bo, out);
}
